// BGATModel_72267119722745
// MI455X (gfx1250) — matrix operand order repaired and checked by reading, not compiled, not run on hardware
//
#include <hip/hip_runtime.h>
#include <hip/hip_bf16.h>

typedef __attribute__((ext_vector_type(16))) _Float16 v16h;
typedef __attribute__((ext_vector_type(8)))  _Float16 v8h;
typedef __attribute__((ext_vector_type(8)))  float    v8f;

#define NBK   5
#define MM    2
#define NN    4
#define NHH   8
#define HDD   4
#define HIDD  32
#define TPB   64                 // 2 waves of 32
#define NWAVE (TPB/32)
#define ROWS_U (TPB*MM)          // 128
#define ROWS_A (TPB*NN)          // 256
#define ROWS   (ROWS_U+ROWS_A)   // 384
#define NTILES (ROWS/16)         // 24
#define UTILES (ROWS_U/16)       // 8
#define ATILES (ROWS_A/16)       // 16

__global__ __launch_bounds__(TPB) void bgat_forward(
    const float* __restrict__ users,  const float* __restrict__ delta_init,
    const float* __restrict__ power_init,
    const float* __restrict__ Wu,     const float* __restrict__ Wa,
    const float* __restrict__ We,     const float* __restrict__ av,
    const float* __restrict__ Wres,
    const float* __restrict__ lnu_g,  const float* __restrict__ lnu_b,
    const float* __restrict__ lna_g,  const float* __restrict__ lna_b,
    const float* __restrict__ mu_w1,  const float* __restrict__ mu_b1,
    const float* __restrict__ mu_w2,  const float* __restrict__ mu_b2,
    const float* __restrict__ ma_w1,  const float* __restrict__ ma_b1,
    const float* __restrict__ ma_w2,  const float* __restrict__ ma_b2,
    const float* __restrict__ rd_w1,  const float* __restrict__ rd_b1,
    const float* __restrict__ rd_w2,  const float* __restrict__ rd_b2,
    const float* __restrict__ rp_w1,  const float* __restrict__ rp_b1,
    const float* __restrict__ rp_w2,  const float* __restrict__ rp_b2,
    float* __restrict__ out, int Btot)
{
    __shared__ __align__(16) _Float16 rowbuf[ROWS * HIDD];   // pre-LN / post-LN rows (f16)
    __shared__ __align__(16) _Float16 hidbuf[ROWS * 16];     // MLP hidden (post-ReLU, f16)
    __shared__ float wWu[64], wWa[64], wWe[32], wav[32], wWres[64];
    __shared__ float wlnug[32], wlnub[32], wlnag[32], wlnab[32];

    const int tid  = threadIdx.x;
    const int lane = tid & 31;
    // wave id, forced into an SGPR so tile loops get scalar control flow
    const int wv   = __builtin_amdgcn_readfirstlane(tid >> 5);

    const int bglob = blockIdx.x * TPB + tid;
    const bool vld  = (bglob < Btot);
    const int bi    = vld ? bglob : (Btot - 1);

    const float Dc = 50.0f, HZc = 5.0f, DMINc = 0.025f, PMAXc = 1.0f;
    const float SCALEc = 0.5f;                         // 1/sqrt(HD=4)
    const float BMAXc  = 2.0f * Dc - (NN - 1) * DMINc; // 99.925

    // persistent per-item state
    float uf[MM][2], af[NN][2];
    #pragma unroll
    for (int m = 0; m < MM; ++m) {
        uf[m][0] = users[(bi * MM + m) * 2 + 0];
        uf[m][1] = users[(bi * MM + m) * 2 + 1];
    }
    #pragma unroll
    for (int n = 0; n < NN; ++n) {
        af[n][0] = power_init[bi * NN + n];
        af[n][1] = delta_init[bi * NN + n];
    }

    for (int d = 0; d < NBK; ++d) {
        __syncthreads();  // previous layer's LDS consumers done
        // ---- stage per-layer broadcast weights into LDS ----
        {
            const int b64 = d * 64;
            wWu[tid]   = Wu[b64 + tid];
            wWa[tid]   = Wa[b64 + tid];
            wWres[tid] = Wres[b64 + tid];
            if (tid < 32) {
                const int b32 = d * 32;
                wWe[tid]   = We[b32 + tid];
                wav[tid]   = av[b32 + tid];
                wlnug[tid] = lnu_g[b32 + tid];
                wlnub[tid] = lnu_b[b32 + tid];
                wlnag[tid] = lna_g[b32 + tid];
                wlnab[tid] = lna_b[b32 + tid];
            }
        }
        __syncthreads();

        // ---- scalar attention (per batch item) ----
        float xpos[NN];
        {
            float da[NN], sd = 1e-6f;
            #pragma unroll
            for (int n = 0; n < NN; ++n) { da[n] = fmaxf(af[n][1], 0.0f); sd += da[n]; }
            float cs = 0.0f;
            #pragma unroll
            for (int n = 0; n < NN; ++n) {
                cs += BMAXc * da[n] / sd;
                xpos[n] = cs - Dc + DMINc * (float)n;
            }
        }
        float edge[MM][NN];
        #pragma unroll
        for (int m = 0; m < MM; ++m)
            #pragma unroll
            for (int n = 0; n < NN; ++n) {
                float dx = uf[m][0] - xpos[n];
                float dy = uf[m][1];
                edge[m][n] = sqrtf(dx * dx + dy * dy);
            }

        for (int k = 0; k < NHH; ++k) {
            float U[MM][HDD], Av[NN][HDD];
            #pragma unroll
            for (int h = 0; h < HDD; ++h) {
                const float w0 = wWu[(k * HDD + h) * 2 + 0];
                const float w1 = wWu[(k * HDD + h) * 2 + 1];
                #pragma unroll
                for (int m = 0; m < MM; ++m) U[m][h] = uf[m][0] * w0 + uf[m][1] * w1;
                const float a0 = wWa[(k * HDD + h) * 2 + 0];
                const float a1 = wWa[(k * HDD + h) * 2 + 1];
                #pragma unroll
                for (int n = 0; n < NN; ++n) Av[n][h] = af[n][0] * a0 + af[n][1] * a1;
            }
            float alpha[MM][NN];
            #pragma unroll
            for (int m = 0; m < MM; ++m) {
                float sc[NN];
                #pragma unroll
                for (int n = 0; n < NN; ++n) {
                    float acc = 0.0f;
                    #pragma unroll
                    for (int h = 0; h < HDD; ++h) {
                        float s = U[m][h] + Av[n][h] + edge[m][n] * wWe[k * HDD + h];
                        s = (s > 0.0f) ? s : 0.2f * s;
                        acc += s * wav[k * HDD + h];
                    }
                    sc[n] = acc * SCALEc;
                }
                float mx = fmaxf(fmaxf(sc[0], sc[1]), fmaxf(sc[2], sc[3]));
                float se = 0.0f;
                #pragma unroll
                for (int n = 0; n < NN; ++n) { sc[n] = expf(sc[n] - mx); se += sc[n]; }
                const float inv = 1.0f / se;
                #pragma unroll
                for (int n = 0; n < NN; ++n) alpha[m][n] = sc[n] * inv;
            }
            // uo -> u-rows (cols 4k..4k+3), plus residual uf @ Wres^T
            #pragma unroll
            for (int m = 0; m < MM; ++m) {
                #pragma unroll
                for (int h = 0; h < HDD; ++h) {
                    float acc = 0.0f;
                    #pragma unroll
                    for (int n = 0; n < NN; ++n)
                        acc += alpha[m][n] * (Av[n][h] + edge[m][n] * wWe[k * HDD + h]);
                    const int j = k * HDD + h;
                    acc += uf[m][0] * wWres[j * 2 + 0] + uf[m][1] * wWres[j * 2 + 1];
                    rowbuf[(tid * MM + m) * HIDD + j] = (_Float16)acc;
                }
            }
            // ao -> a-rows
            #pragma unroll
            for (int n = 0; n < NN; ++n) {
                #pragma unroll
                for (int h = 0; h < HDD; ++h) {
                    float acc = 0.0f;
                    #pragma unroll
                    for (int m = 0; m < MM; ++m)
                        acc += alpha[m][n] * (U[m][h] + edge[m][n] * wWe[k * HDD + h]);
                    rowbuf[(ROWS_U + tid * NN + n) * HIDD + k * HDD + h] = (_Float16)acc;
                }
            }
        }

        // ---- LayerNorm rows in place (each thread owns its rows) ----
        #pragma unroll
        for (int m = 0; m < MM; ++m) {
            _Float16* p = &rowbuf[(tid * MM + m) * HIDD];
            float x[HIDD], mu = 0.0f;
            #pragma unroll
            for (int j = 0; j < HIDD; ++j) { x[j] = (float)p[j]; mu += x[j]; }
            mu *= (1.0f / HIDD);
            float v = 0.0f;
            #pragma unroll
            for (int j = 0; j < HIDD; ++j) { float t = x[j] - mu; v += t * t; }
            const float rs = rsqrtf(v * (1.0f / HIDD) + 1e-5f);
            #pragma unroll
            for (int j = 0; j < HIDD; ++j)
                p[j] = (_Float16)((x[j] - mu) * rs * wlnug[j] + wlnub[j]);
        }
        #pragma unroll
        for (int n = 0; n < NN; ++n) {
            _Float16* p = &rowbuf[(ROWS_U + tid * NN + n) * HIDD];
            float x[HIDD], mu = 0.0f;
            #pragma unroll
            for (int j = 0; j < HIDD; ++j) { x[j] = (float)p[j]; mu += x[j]; }
            mu *= (1.0f / HIDD);
            float v = 0.0f;
            #pragma unroll
            for (int j = 0; j < HIDD; ++j) { float t = x[j] - mu; v += t * t; }
            const float rs = rsqrtf(v * (1.0f / HIDD) + 1e-5f);
            #pragma unroll
            for (int j = 0; j < HIDD; ++j)
                p[j] = (_Float16)((x[j] - mu) * rs * wlnag[j] + wlnab[j]);
        }
        __syncthreads();

        // ---- WMMA: hidden = relu( LN_rows @ w1^T + b1 ), K=32 exactly ----
        {
            const int ncol  = lane & 15;
            const int kb    = (lane >> 4) * 16;  // unused; B operand uses khalf as its K base too
            const int khalf = (lane >> 4) * 8;   // A-operand K base (halves)
            const int arow0 = lane & 15;         // A-operand row within tile
            const int rofs  = (lane >> 4) << 3;  // D-operand row offset within tile
            const int col   = lane & 15;         // D-operand column

            v16h bu, ba;
            #pragma unroll
            for (int j = 0; j < 16; ++j) {
                bu[j] = (_Float16)mu_w1[d * 512 + ncol * 32 + khalf + (j & 7) + ((j >> 3) << 4)];
                ba[j] = (_Float16)ma_w1[d * 512 + ncol * 32 + khalf + (j & 7) + ((j >> 3) << 4)];
            }
            const float bias_u = mu_b1[d * 16 + ncol];
            const float bias_a = ma_b1[d * 16 + ncol];

            // u tiles: t = wv, wv+NWAVE, ... < UTILES   (compile-time trip count)
            #pragma unroll
            for (int i = 0; i < UTILES / NWAVE; ++i) {
                const int t    = wv + i * NWAVE;
                const int arow = t * 16 + arow0;
                v8h a0 = *(const v8h*)&rowbuf[arow * HIDD + khalf];
                v8h a1 = *(const v8h*)&rowbuf[arow * HIDD + khalf + 16];
                v16h aM;
                #pragma unroll
                for (int j = 0; j < 8; ++j) { aM[j] = a0[j]; aM[8 + j] = a1[j]; }
                v8f c = {};
                c = __builtin_amdgcn_wmma_f32_16x16x32_f16(
                        false, aM, false, bu, (short)0, c, false, false);
                const int rbase = t * 16 + rofs;
                #pragma unroll
                for (int r = 0; r < 8; ++r)
                    hidbuf[(rbase + r) * 16 + col] = (_Float16)fmaxf(c[r] + bias_u, 0.0f);
            }
            // a tiles: t = UTILES + wv + i*NWAVE
            #pragma unroll
            for (int i = 0; i < ATILES / NWAVE; ++i) {
                const int t    = UTILES + wv + i * NWAVE;
                const int arow = t * 16 + arow0;
                v8h a0 = *(const v8h*)&rowbuf[arow * HIDD + khalf];
                v8h a1 = *(const v8h*)&rowbuf[arow * HIDD + khalf + 16];
                v16h aM;
                #pragma unroll
                for (int j = 0; j < 8; ++j) { aM[j] = a0[j]; aM[8 + j] = a1[j]; }
                v8f c = {};
                c = __builtin_amdgcn_wmma_f32_16x16x32_f16(
                        false, aM, false, ba, (short)0, c, false, false);
                const int rbase = t * 16 + rofs;
                #pragma unroll
                for (int r = 0; r < 8; ++r)
                    hidbuf[(rbase + r) * 16 + col] = (_Float16)fmaxf(c[r] + bias_a, 0.0f);
            }
        }
        __syncthreads();

        // ---- second MLP layer (16 -> 2), scalar, updates loop state ----
        #pragma unroll
        for (int m = 0; m < MM; ++m) {
            const _Float16* hp = &hidbuf[(tid * MM + m) * 16];
            float h[16];
            #pragma unroll
            for (int j = 0; j < 16; ++j) h[j] = (float)hp[j];
            #pragma unroll
            for (int c2 = 0; c2 < 2; ++c2) {
                float acc = mu_b2[d * 2 + c2];
                #pragma unroll
                for (int j = 0; j < 16; ++j) acc += h[j] * mu_w2[d * 32 + c2 * 16 + j];
                uf[m][c2] = acc;
            }
        }
        #pragma unroll
        for (int n = 0; n < NN; ++n) {
            const _Float16* hp = &hidbuf[(ROWS_U + tid * NN + n) * 16];
            float h[16];
            #pragma unroll
            for (int j = 0; j < 16; ++j) h[j] = (float)hp[j];
            #pragma unroll
            for (int c2 = 0; c2 < 2; ++c2) {
                float acc = ma_b2[d * 2 + c2];
                #pragma unroll
                for (int j = 0; j < 16; ++j) acc += h[j] * ma_w2[d * 32 + c2 * 16 + j];
                af[n][c2] = acc;
            }
        }
    }

    // ---- output heads ----
    float rawd[NN], rawp[NN];
    {
        float t1[2 * NN];
        #pragma unroll
        for (int i = 0; i < 2 * NN; ++i) {
            float acc = rd_b1[i];
            #pragma unroll
            for (int n = 0; n < NN; ++n) acc += af[n][1] * rd_w1[i * NN + n];
            t1[i] = fmaxf(acc, 0.0f);
        }
        #pragma unroll
        for (int n = 0; n < NN; ++n) {
            float acc = rd_b2[n];
            #pragma unroll
            for (int i = 0; i < 2 * NN; ++i) acc += t1[i] * rd_w2[n * 2 * NN + i];
            rawd[n] = fmaxf(acc, 0.001f);
        }
        float t2[2 * NN];
        #pragma unroll
        for (int i = 0; i < 2 * NN; ++i) {
            float acc = rp_b1[i];
            #pragma unroll
            for (int n = 0; n < NN; ++n) acc += af[n][0] * rp_w1[i * NN + n];
            t2[i] = fmaxf(acc, 0.0f);
        }
        #pragma unroll
        for (int n = 0; n < NN; ++n) {
            float acc = rp_b2[n];
            #pragma unroll
            for (int i = 0; i < 2 * NN; ++i) acc += t2[i] * rp_w2[n * 2 * NN + i];
            rawp[n] = fmaxf(acc, 0.001f);
        }
    }
    float sumd = rawd[0] + rawd[1] + rawd[2] + rawd[3];
    const float sden = BMAXc / sumd;
    float sdel[NN], xn[NN];
    {
        float cs = 0.0f;
        #pragma unroll
        for (int n = 0; n < NN; ++n) {
            sdel[n] = sden * rawd[n];
            cs += sdel[n];
            xn[n] = cs + DMINc * (float)n - Dc * (float)(n + 1);
        }
    }
    float sump = rawp[0] + rawp[1] + rawp[2] + rawp[3] + 1e-6f;
    const float pscl = PMAXc / fmaxf(PMAXc, sump);

    if (vld) {
        const size_t B4 = (size_t)Btot * NN;
        #pragma unroll
        for (int n = 0; n < NN; ++n) {
            out[(size_t)bglob * NN + n]      = pscl * rawp[n];   // scaled_power
            out[B4 + (size_t)bglob * NN + n] = sdel[n];          // scaled_delta
            const size_t pb = 2 * B4 + ((size_t)bglob * NN + n) * 3;
            out[pb + 0] = xn[n];
            out[pb + 1] = 0.0f;
            out[pb + 2] = HZc;                                   // final_positions
        }
    }
}

extern "C" void kernel_launch(void* const* d_in, const int* in_sizes, int n_in,
                              void* d_out, int out_size, void* d_ws, size_t ws_size,
                              hipStream_t stream) {
    const int B = in_sizes[0] / 4;   // users is (B, 2, 2)
    const int nblk = (B + TPB - 1) / TPB;
    bgat_forward<<<nblk, TPB, 0, stream>>>(
        (const float*)d_in[0],  (const float*)d_in[1],  (const float*)d_in[2],
        (const float*)d_in[3],  (const float*)d_in[4],  (const float*)d_in[5],
        (const float*)d_in[6],  (const float*)d_in[7],  (const float*)d_in[8],
        (const float*)d_in[9],  (const float*)d_in[10], (const float*)d_in[11],
        (const float*)d_in[12], (const float*)d_in[13], (const float*)d_in[14],
        (const float*)d_in[15], (const float*)d_in[16], (const float*)d_in[17],
        (const float*)d_in[18], (const float*)d_in[19], (const float*)d_in[20],
        (const float*)d_in[21], (const float*)d_in[22], (const float*)d_in[23],
        (const float*)d_in[24], (const float*)d_in[25], (const float*)d_in[26],
        (const float*)d_in[27],
        (float*)d_out, B);
}
